// E31SparseGatedCell_63995012710825
// MI455X (gfx1250) — matrix operand order repaired and checked by reading, not compiled, not run on hardware
//
#include <hip/hip_runtime.h>
#include <hip/hip_bf16.h>

#define T_LEN 512
#define B_SZ  16
#define D_DIM 1024
#define NWG_REC 64   // 64 workgroups x 16 columns = 1024 output columns

typedef __bf16 bf16_t;
typedef __bf16 v8bf  __attribute__((ext_vector_type(8)));
typedef __bf16 v16bf __attribute__((ext_vector_type(16)));
typedef float  v8f   __attribute__((ext_vector_type(8)));

static __device__ __forceinline__ v16bf cat8(v8bf lo, v8bf hi) {
  return __builtin_shufflevector(lo, hi, 0,1,2,3,4,5,6,7,8,9,10,11,12,13,14,15);
}

// ---------------------------------------------------------------------------
// Kernel 0: init — h0 -> d_out h-row0 (fp32) and hbuf[0] (bf16); zero barrier.
// ---------------------------------------------------------------------------
__global__ __launch_bounds__(256) void k_init(const float* __restrict__ h0,
                                              float* __restrict__ out_h,
                                              bf16_t* __restrict__ hbuf,
                                              unsigned* __restrict__ bar) {
  int i = blockIdx.x * blockDim.x + threadIdx.x;
  if (i < B_SZ * D_DIM) {
    float v = h0[i];
    out_h[i] = v;                 // h[0] = h0
    hbuf[i]  = (bf16_t)v;         // bf16 working copy, buffer parity 0
  }
  if (i < 8) bar[i] = 0u;
}

// ---------------------------------------------------------------------------
// Kernel 0b: one-pass fp32 -> bf16 materialization of x and W_x.
// 8 elements per thread (two float4 loads, one 16B store).
// ---------------------------------------------------------------------------
__global__ __launch_bounds__(256) void k_cvt_bf16(const float* __restrict__ src,
                                                  bf16_t* __restrict__ dst,
                                                  int n8) {   // n/8
  int i = blockIdx.x * blockDim.x + threadIdx.x;
  if (i < n8) {
    const float* p = src + (size_t)i * 8;
    v8bf r;
    #pragma unroll
    for (int j = 0; j < 8; ++j) r[j] = (__bf16)p[j];
    *(v8bf*)(dst + (size_t)i * 8) = r;
  }
}

// ---------------------------------------------------------------------------
// Kernel 1: 1.5-entmax gate over rows of z. One block (256 thr) per row.
// Bitonic sort (descending) + inclusive scan in LDS, per the reference algo.
// ---------------------------------------------------------------------------
__global__ __launch_bounds__(256) void k_entmax(const float* __restrict__ z,
                                                float* __restrict__ gate) {
  __shared__ float s_val[D_DIM];
  __shared__ float s_cum[D_DIM];
  __shared__ float s_red[256];
  const int row = blockIdx.x;
  const int tid = threadIdx.x;
  const float* zr = z + (size_t)row * D_DIM;

  float v[4];
  float mx = -INFINITY;
  #pragma unroll
  for (int ii = 0; ii < 4; ++ii) { v[ii] = zr[tid + 256 * ii]; mx = fmaxf(mx, v[ii]); }
  s_red[tid] = mx; __syncthreads();
  for (int off = 128; off > 0; off >>= 1) {
    if (tid < off) s_red[tid] = fmaxf(s_red[tid], s_red[tid + off]);
    __syncthreads();
  }
  mx = s_red[0]; __syncthreads();
  #pragma unroll
  for (int ii = 0; ii < 4; ++ii) { v[ii] -= mx; s_val[tid + 256 * ii] = v[ii]; }
  __syncthreads();

  // bitonic sort, descending
  for (int k = 2; k <= D_DIM; k <<= 1) {
    for (int j = k >> 1; j > 0; j >>= 1) {
      #pragma unroll
      for (int ii = 0; ii < 4; ++ii) {
        int i = tid + 256 * ii;
        int ixj = i ^ j;
        if (ixj > i) {
          float a = s_val[i], b = s_val[ixj];
          bool desc = ((i & k) == 0);
          if (desc ? (a < b) : (a > b)) { s_val[i] = b; s_val[ixj] = a; }
        }
      }
      __syncthreads();
    }
  }

  // inclusive scan of sorted values
  #pragma unroll
  for (int ii = 0; ii < 4; ++ii) { int i = tid + 256 * ii; s_cum[i] = s_val[i]; }
  __syncthreads();
  for (int off = 1; off < D_DIM; off <<= 1) {
    float tprev[4];
    #pragma unroll
    for (int ii = 0; ii < 4; ++ii) {
      int i = tid + 256 * ii;
      tprev[ii] = (i >= off) ? s_cum[i - off] : 0.0f;
    }
    __syncthreads();
    #pragma unroll
    for (int ii = 0; ii < 4; ++ii) { int i = tid + 256 * ii; s_cum[i] += tprev[ii]; }
    __syncthreads();
  }

  // support count -> k_star
  int cnt = 0;
  #pragma unroll
  for (int ii = 0; ii < 4; ++ii) {
    int i = tid + 256 * ii;
    float tau = (s_cum[i] - 1.0f) / (float)(i + 1);
    cnt += (s_val[i] >= tau) ? 1 : 0;
  }
  s_red[tid] = (float)cnt; __syncthreads();
  for (int off = 128; off > 0; off >>= 1) {
    if (tid < off) s_red[tid] += s_red[tid + off];
    __syncthreads();
  }
  int kstar = (int)s_red[0];
  if (kstar < 1) kstar = 1;
  float tau_star = (s_cum[kstar - 1] - 1.0f) / (float)kstar;
  __syncthreads();

  // unnormalized sqrt weights + normalization
  float p[4];
  float ps = 0.0f;
  #pragma unroll
  for (int ii = 0; ii < 4; ++ii) {
    p[ii] = sqrtf(fmaxf(v[ii] - tau_star, 0.0f));
    ps += p[ii];
  }
  s_red[tid] = ps; __syncthreads();
  for (int off = 128; off > 0; off >>= 1) {
    if (tid < off) s_red[tid] += s_red[tid + off];
    __syncthreads();
  }
  float inv = 1.0f / (s_red[0] + 1e-10f);
  #pragma unroll
  for (int ii = 0; ii < 4; ++ii)
    gate[(size_t)row * D_DIM + tid + 256 * ii] = p[ii] * inv;
}

// ---------------------------------------------------------------------------
// Kernel 2: xw = x @ W_x^T + b  via v_wmma_f32_16x16x32_bf16, pure bf16 loads.
// Grid (M/16, N/64), block 128 (4 waves); wave w owns 16x16 tile at col +16w.
// A layout: lane<16 -> row=lane, K {0..15}; lane>=16 -> K {16..31}.
// B layout: lane -> column (= W_x row), 16 contiguous K per lane.
// ---------------------------------------------------------------------------
__global__ __launch_bounds__(128) void k_gemm_xw(const bf16_t* __restrict__ xb,
                                                 const bf16_t* __restrict__ Wxb,
                                                 const float* __restrict__ bias,
                                                 float* __restrict__ xw) {
  const int tile_m = blockIdx.x;            // 0..511  (rows of x, 16 each)
  const int nb     = blockIdx.y;            // 0..15   (64-col groups)
  const int tid    = threadIdx.x;
  const int wave   = tid >> 5;
  const int lane   = tid & 31;

  const int koff   = (lane < 16) ? 0 : 8;
  const int khalf  = (lane < 16) ? 0 : 16;
  const int colb   = nb * 64 + wave * 16;
  const int rowg   = tile_m * 16 + (lane & 15);  // A row for this lane
  const int ncol   = colb + (lane & 15);         // B column / output column

  const bf16_t* arow = xb  + (size_t)rowg * D_DIM;
  const bf16_t* brow = Wxb + (size_t)ncol * D_DIM;

  v8f acc = {};
  #pragma unroll 4
  for (int kc = 0; kc < D_DIM; kc += 32) {
    v8bf a0 = *(const v8bf*)(arow + kc + khalf);
    v8bf a1 = *(const v8bf*)(arow + kc + 8 + khalf);
    v16bf A = cat8(a0, a1);
    v16bf B = *(const v16bf*)(brow + kc + khalf);
    acc = __builtin_amdgcn_wmma_f32_16x16x32_bf16(false, A, false, B,
                                                  (short)0, acc, false, false);
  }

  const float bb = bias[ncol];
  #pragma unroll
  for (int r = 0; r < 8; ++r) {
    int M = r + ((lane >> 4) << 3);
    xw[(size_t)(tile_m * 16 + M) * D_DIM + ncol] = acc[r] + bb;
  }
}

// ---------------------------------------------------------------------------
// Kernel 3: persistent recurrence. 64 single-wave workgroups, each owns a
// 16-column strip of W_h kept in LDS as bf16 for the whole sequence.
// Per step: 32 chained WMMAs (K=1024), fused tanh + gate + stores, grid sync.
// ---------------------------------------------------------------------------
__global__ __launch_bounds__(32) void k_recurrence(const float* __restrict__ Wh,
                                                   const float* __restrict__ xw,
                                                   const float* __restrict__ gate,
                                                   bf16_t* __restrict__ hbuf,   // 2 x [16,1024]
                                                   float* __restrict__ out_main,
                                                   float* __restrict__ out_h,
                                                   unsigned* __restrict__ bar) {
  __shared__ __align__(16) bf16_t s_wh[16 * D_DIM];   // 32 KB strip of W_h

  const int g     = blockIdx.x;           // 0..63
  const int colb  = g * 16;
  const int lane  = threadIdx.x;          // single wave32

  // Load + convert this block's 16 rows of W_h once (resident across steps).
  for (int i = lane; i < 16 * D_DIM; i += 32) {
    int r = i >> 10, c = i & (D_DIM - 1);
    s_wh[i] = (bf16_t)Wh[(size_t)(colb + r) * D_DIM + c];
  }
  __syncthreads();

  const int m     = lane & 15;
  const int koff  = (lane < 16) ? 0 : 8;
  const int khalf = (lane < 16) ? 0 : 16;
  const int col   = colb + (lane & 15);

  unsigned* cnt = bar + 0;
  unsigned* gen = bar + 1;

  for (int t = 0; t < T_LEN; ++t) {
    const bf16_t* hp = hbuf + (size_t)(t & 1) * (B_SZ * D_DIM);
    bf16_t*       hn = hbuf + (size_t)((t + 1) & 1) * (B_SZ * D_DIM);

    // prefetch next step's xw/gate rows for this strip
    if (t + 1 < T_LEN && lane < 16) {
      __builtin_prefetch(xw   + (size_t)((t + 1) * B_SZ + lane) * D_DIM + colb, 0, 1);
      __builtin_prefetch(gate + (size_t)((t + 1) * B_SZ + lane) * D_DIM + colb, 0, 1);
    }

    v8f acc = {};
    #pragma unroll 4
    for (int kc = 0; kc < D_DIM; kc += 32) {
      v8bf a0 = *(const v8bf*)(hp + (size_t)m * D_DIM + kc + khalf);
      v8bf a1 = *(const v8bf*)(hp + (size_t)m * D_DIM + kc + 8 + khalf);
      v16bf A = cat8(a0, a1);
      v16bf B = *(const v16bf*)(s_wh + (size_t)(lane & 15) * D_DIM + kc + khalf);
      acc = __builtin_amdgcn_wmma_f32_16x16x32_bf16(false, A, false, B,
                                                    (short)0, acc, false, false);
    }

    // fused epilogue: tanh, h (fp32 + bf16), gated output
    #pragma unroll
    for (int r = 0; r < 8; ++r) {
      int M = r + ((lane >> 4) << 3);
      size_t idx = (size_t)(t * B_SZ + M) * D_DIM + col;
      float v = tanhf(acc[r] + xw[idx]);
      out_h[(size_t)((t + 1) * B_SZ + M) * D_DIM + col] = v;
      hn[(size_t)M * D_DIM + col] = (bf16_t)v;
      out_main[idx] = v * gate[idx];
    }

    // grid barrier (generation counter, deterministic, self-resetting)
    __threadfence();
    if (lane == 0) {
      unsigned a = __hip_atomic_fetch_add(cnt, 1u, __ATOMIC_ACQ_REL,
                                          __HIP_MEMORY_SCOPE_AGENT);
      if (a == (unsigned)(NWG_REC - 1)) {
        __hip_atomic_store(cnt, 0u, __ATOMIC_RELAXED, __HIP_MEMORY_SCOPE_AGENT);
        __hip_atomic_fetch_add(gen, 1u, __ATOMIC_ACQ_REL, __HIP_MEMORY_SCOPE_AGENT);
      } else {
        while (__hip_atomic_load(gen, __ATOMIC_ACQUIRE, __HIP_MEMORY_SCOPE_AGENT)
               <= (unsigned)t) {
          __builtin_amdgcn_s_sleep(1);
        }
      }
    }
    __syncthreads();
  }
}

// ---------------------------------------------------------------------------
extern "C" void kernel_launch(void* const* d_in, const int* in_sizes, int n_in,
                              void* d_out, int out_size, void* d_ws, size_t ws_size,
                              hipStream_t stream) {
  const float* x  = (const float*)d_in[0];   // [T,B,D]
  const float* z  = (const float*)d_in[1];   // [T,B,D]
  const float* h0 = (const float*)d_in[2];   // [B,D]
  const float* Wx = (const float*)d_in[3];   // [D,D]
  const float* Wh = (const float*)d_in[4];   // [D,D]
  const float* b  = (const float*)d_in[5];   // [D]
  (void)in_sizes; (void)n_in; (void)out_size; (void)ws_size;

  float* out      = (float*)d_out;
  float* out_main = out;                                    // [T,B,D]
  float* out_h    = out + (size_t)T_LEN * B_SZ * D_DIM;     // [T+1,B,D]

  const size_t nTBD = (size_t)T_LEN * B_SZ * D_DIM;         // 8,388,608
  const size_t nDD  = (size_t)D_DIM * D_DIM;                // 1,048,576
  float*    gate  = (float*)d_ws;
  float*    xw    = gate + nTBD;
  bf16_t*   hbuf  = (bf16_t*)(xw + nTBD);                   // 2 x [B,D] bf16
  unsigned* bar   = (unsigned*)(hbuf + 2 * B_SZ * D_DIM);
  bf16_t*   x_bf  = (bf16_t*)(bar + 8);                     // [T,B,D] bf16
  bf16_t*   Wx_bf = x_bf + nTBD;                            // [D,D] bf16

  k_init<<<64, 256, 0, stream>>>(h0, out_h, hbuf, bar);
  k_cvt_bf16<<<(int)((nTBD / 8 + 255) / 256), 256, 0, stream>>>(x, x_bf, (int)(nTBD / 8));
  k_cvt_bf16<<<(int)((nDD / 8 + 255) / 256), 256, 0, stream>>>(Wx, Wx_bf, (int)(nDD / 8));
  k_entmax<<<T_LEN * B_SZ, 256, 0, stream>>>(z, gate);
  k_gemm_xw<<<dim3(T_LEN * B_SZ / 16, D_DIM / 64), 128, 0, stream>>>(x_bf, Wx_bf, b, xw);
  k_recurrence<<<NWG_REC, 32, 0, stream>>>(Wh, xw, gate, hbuf, out_main, out_h, bar);
}
